// SpikingAutoencoder_57844619542919
// MI455X (gfx1250) — matrix operand order repaired and checked by reading, not compiled, not run on hardware
//
#include <hip/hip_runtime.h>
#include <hip/hip_bf16.h>
#include <stddef.h>

// ---------------------------------------------------------------------------
// Spiking autoencoder, CDNA5 (gfx1250) persistent-state WMMA implementation.
//
//   T=100, B=256, D=784, H=256, THR=1.0, BETA=0.9
//
//   grid = 16 workgroups (one per 16-row batch tile), 512 threads = 16 waves.
//   Each wave owns one N-tile of layer 1 (H=256 -> 16 tiles) and 3-4 N-tiles
//   of layer 2 (D=784 -> 49 tiles). Membrane potentials live in v8f WMMA
//   accumulator fragments for the whole 100-step loop. Packed f16 weights
//   (~0.8 MB) are re-streamed from L2 each step (16 MACs reuse per element);
//   K-loops use bounded unrolling so the compiler cannot hoist the weight
//   loads across the time loop and spill them to scratch.
// ---------------------------------------------------------------------------

typedef __attribute__((ext_vector_type(16))) _Float16 v16h;
typedef __attribute__((ext_vector_type(8)))  _Float16 v8h;
typedef __attribute__((ext_vector_type(4)))  _Float16 v4h;
typedef __attribute__((ext_vector_type(8)))  float    v8f;
typedef __attribute__((ext_vector_type(4)))  float    v4f;

#define T_STEPS 100
#define BATCH   256
#define DIN     784
#define HID     256
#define KP1     800           // DIN padded to multiple of 32
#define KT1     (KP1 / 32)    // 25 K-tiles for GEMM1
#define KT2     (HID / 32)    // 8  K-tiles for GEMM2
#define NT1     (HID / 16)    // 16 N-tiles layer 1
#define NT2     ((DIN + 15) / 16) // 49 N-tiles layer 2
#define THR_F   1.0f
#define BETA_F  0.9f

#define W1P_ELEMS ((size_t)KT1 * NT1 * 32 * 16)   // 204800 halves
#define W2P_ELEMS ((size_t)KT2 * NT2 * 32 * 16)   // 200704 halves

// ---------------------------------------------------------------------------
// Pack fp32 weight matrix W[N][K] (we need B[k][n] = W[n][k]) into the
// per-lane V_WMMA_F32_16X16X32_F16 B-fragment layout:
//   fragment (kt, nt): lane L holds 16 contiguous f16 = B[kt*32 + (e>>3)*16 + (L>>4)*8 + (e&7),
//                                                        nt*16 + (L&15)], e=0..15
// so the main kernel fetches each B fragment with two global_load_b128s.
// ---------------------------------------------------------------------------
__global__ void snn_pack_w(const float* __restrict__ W, _Float16* __restrict__ out,
                           int N, int K, int nTiles, int kTiles) {
    int idx   = blockIdx.x * blockDim.x + threadIdx.x;
    int total = kTiles * nTiles * 32 * 16;
    if (idx >= total) return;
    int e    = idx & 15;
    int lane = (idx >> 4) & 31;
    int nt   = (idx >> 9) % nTiles;
    int kt   = (idx >> 9) / nTiles;
    int n = nt * 16 + (lane & 15);
    int k = kt * 32 + (e >> 3) * 16 + (lane >> 4) * 8 + (e & 7);
    float v = (k < K && n < N) ? W[(size_t)n * K + k] : 0.0f;
    out[idx] = (_Float16)v;
}

// A-fragment (16x32 f16) loader from a row-major LDS tile.
// Lane L: M = L&15, K-half = L>>4; ISA layout = two runs of 8 consecutive K:
//   v16h[0..7]  = K(kBase + (L>>4)*8 .. +7)        -> VGPRs 0-3
//   v16h[8..15] = K(kBase + 16 + (L>>4)*8 .. +7)   -> VGPRs 4-7
__device__ __forceinline__ v16h load_a_frag(const _Float16* base, int rowStride, int kBase) {
    int lane = threadIdx.x & 31;
    int m    = lane & 15;
    int kh   = lane >> 4;
    const _Float16* p = base + (size_t)m * rowStride + kBase + kh * 8;
    v8h lo = *(const v8h*)(p);
    v8h hi = *(const v8h*)(p + 16);
    v16h r;
#pragma unroll
    for (int i = 0; i < 8; ++i) { r[i] = lo[i]; r[i + 8] = hi[i]; }
    return r;
}

__global__ __launch_bounds__(512, 1) void snn_persistent(
        const float* __restrict__ spikes,   // [T, B, D]
        const float* __restrict__ b1,       // [H]
        const float* __restrict__ b2,       // [D]
        const _Float16* __restrict__ w1p,   // packed B-fragments, layer 1
        const _Float16* __restrict__ w2p,   // packed B-fragments, layer 2
        float* __restrict__ out) {          // [T, B, D]

    __shared__ __align__(16) _Float16 xT[16 * KP1];   // input tile, A layout (25.6 KB)
    __shared__ __align__(16) _Float16 s1[16 * HID];   // layer-1 spikes, A layout (8 KB)

    const int tid   = threadIdx.x;
    const int wave  = tid >> 5;       // 0..15
    const int lane  = tid & 31;
    const int laneN = lane & 15;      // N index within C fragment
    const int laneM = (lane >> 4) * 8;// M base within C fragment (VGPR r -> M = r + laneM)
    const int bBase = blockIdx.x * 16;

    // Zero the whole x tile once: cols [784, 800) stay zero forever (K padding).
    for (int i = tid; i < 16 * KP1; i += 512) xT[i] = (_Float16)0.0f;

    // Persistent LIF state in WMMA C-fragment layout.
    v8f mem1 = {};
    v8f mem2[4] = {{}, {}, {}, {}};

    // Biases for the columns this lane accumulates.
    const float bias1 = b1[wave * 16 + laneN];
    float bias2[4] = {0.f, 0.f, 0.f, 0.f};
#pragma unroll
    for (int j = 0; j < 4; ++j) {
        int nt = wave + 16 * j;
        if (nt < NT2) bias2[j] = b2[nt * 16 + laneN];
    }

    __syncthreads();

    for (int t = 0; t < T_STEPS; ++t) {
        // ---- stage x_t tile: fp32 -> f16, row-major [16][800] in LDS ----
        // Wave w owns row w; v4f streaming (non-temporal) loads,
        // packed 4xf16 ds_store_b64 stores. 784 = 6*128 + 16 floats.
        {
            const v4f* src4 = (const v4f*)(spikes +
                              ((size_t)t * BATCH + bBase + wave) * DIN);
            _Float16* dst = xT + wave * KP1;
#pragma unroll
            for (int i = 0; i < 6; ++i) {
                int c4 = lane + i * 32;
                v4f v = __builtin_nontemporal_load(src4 + c4);
                v4h h; h[0] = (_Float16)v[0]; h[1] = (_Float16)v[1];
                       h[2] = (_Float16)v[2]; h[3] = (_Float16)v[3];
                *(v4h*)(dst + c4 * 4) = h;
            }
            if (lane < 4) {
                int c4 = 192 + lane;
                v4f v = __builtin_nontemporal_load(src4 + c4);
                v4h h; h[0] = (_Float16)v[0]; h[1] = (_Float16)v[1];
                       h[2] = (_Float16)v[2]; h[3] = (_Float16)v[3];
                *(v4h*)(dst + c4 * 4) = h;
            }
        }
        __syncthreads();

        // ---- GEMM1: cur1 tile [16 x 16], N-tile = wave ----
        // Bounded unroll: keeps B-fragment addresses loop-variant so they are
        // re-streamed from L2 instead of being hoisted + spilled to scratch.
        v8f acc1 = {};
#pragma unroll 5
        for (int kt = 0; kt < KT1; ++kt) {
            v16h a = load_a_frag(xT, KP1, kt * 32);
            v16h b = *(const v16h*)(w1p + ((size_t)(kt * NT1 + wave) * 32 + lane) * 16);
            acc1 = __builtin_amdgcn_wmma_f32_16x16x32_f16(
                       false, a, false, b, (short)0, acc1, false, false);
        }

        // ---- LIF layer 1 (reset-by-subtraction), emit spk1 to LDS ----
#pragma unroll
        for (int r = 0; r < 8; ++r) {
            float m     = mem1[r];
            float reset = (m > THR_F) ? THR_F : 0.0f;
            m = BETA_F * m + (acc1[r] + bias1) - reset;
            mem1[r] = m;
            float s = (m > THR_F) ? 1.0f : 0.0f;
            s1[(r + laneM) * HID + wave * 16 + laneN] = (_Float16)s;
        }

        // Prefetch next timestep's input rows while GEMM2 runs.
        if (t + 1 < T_STEPS) {
            const float* nx = spikes + ((size_t)(t + 1) * BATCH + bBase + wave) * DIN;
            __builtin_prefetch(nx + lane * 16, 0, 1);
        }
        __syncthreads();

        // ---- GEMM2 + LIF layer 2: N-tiles {wave, wave+16, wave+32, wave+48<49} ----
#pragma unroll
        for (int j = 0; j < 4; ++j) {
            int nt = wave + 16 * j;
            if (nt >= NT2) break;                 // wave-uniform: EXEC stays full
            v8f acc2 = {};
#pragma unroll 4
            for (int kt = 0; kt < KT2; ++kt) {
                v16h a = load_a_frag(s1, HID, kt * 32);
                v16h b = *(const v16h*)(w2p + ((size_t)(kt * NT2 + nt) * 32 + lane) * 16);
                acc2 = __builtin_amdgcn_wmma_f32_16x16x32_f16(
                           false, a, false, b, (short)0, acc2, false, false);
            }
#pragma unroll
            for (int r = 0; r < 8; ++r) {
                float m     = mem2[j][r];
                float reset = (m > THR_F) ? THR_F : 0.0f;
                m = BETA_F * m + (acc2[r] + bias2[j]) - reset;
                mem2[j][r] = m;
                float s = (m > THR_F) ? 1.0f : 0.0f;
                __builtin_nontemporal_store(
                    s, out + (size_t)t * BATCH * DIN
                         + (size_t)(bBase + r + laneM) * DIN + nt * 16 + laneN);
            }
        }
        __syncthreads();   // protect xT / s1 before next step's writes
    }
}

extern "C" void kernel_launch(void* const* d_in, const int* in_sizes, int n_in,
                              void* d_out, int out_size, void* d_ws, size_t ws_size,
                              hipStream_t stream) {
    const float* sp = (const float*)d_in[0];   // [100,256,784]
    const float* W1 = (const float*)d_in[1];   // [256,784]
    const float* b1 = (const float*)d_in[2];   // [256]
    const float* W2 = (const float*)d_in[3];   // [784,256]
    const float* b2 = (const float*)d_in[4];   // [784]
    float* out = (float*)d_out;

    _Float16* w1p = (_Float16*)d_ws;
    _Float16* w2p = w1p + W1P_ELEMS;           // ~0.81 MB total in d_ws

    int tot1 = (int)W1P_ELEMS;
    int tot2 = (int)W2P_ELEMS;
    snn_pack_w<<<(tot1 + 255) / 256, 256, 0, stream>>>(W1, w1p, HID, DIN, NT1, KT1);
    snn_pack_w<<<(tot2 + 255) / 256, 256, 0, stream>>>(W2, w2p, DIN, HID, NT2, KT2);

    snn_persistent<<<BATCH / 16, 512, 0, stream>>>(sp, b1, b2, w1p, w2p, out);
}
